// SpatialMambaHybridModel_51067161150073
// MI455X (gfx1250) — compile-verified, matrix operand order checked by reading, not run on hardware
//
#include <hip/hip_runtime.h>
#include <hip/hip_bf16.h>
#include <math.h>

#define BATCH  512
#define SEQ    64
#define DM     64
#define DI     128
#define DS     32
#define XDBL_W 68      // DT_RANK(4) + D_STATE(32) + D_STATE(32)

typedef __attribute__((ext_vector_type(16))) _Float16      v16h;
typedef __attribute__((ext_vector_type(8)))  float         v8f;
typedef __attribute__((ext_vector_type(4)))  unsigned int  v4u;
typedef __attribute__((ext_vector_type(4)))  int           v4i;
typedef __attribute__((ext_vector_type(8)))  int           v8i;

__device__ __forceinline__ float sigmoidf_(float x) { return 1.0f / (1.0f + __expf(-x)); }
__device__ __forceinline__ float siluf_(float x)    { return x * sigmoidf_(x); }
__device__ __forceinline__ float softplusf_(float x){ return fmaxf(x, 0.0f) + log1pf(__expf(-fabsf(x))); }

// Load one 16-lane A/B fragment (f32 -> f16) for v_wmma_f32_16x16x32_f16.
// Row pointer p, k-window base b0 (already includes lane-half offset).
// K layout per lane-half: {b0..b0+7, b0+16..b0+23}.
__device__ __forceinline__ v16h load_frag_(const float* __restrict__ p, int b0)
{
    float4 a0 = *(const float4*)(p + b0);
    float4 a1 = *(const float4*)(p + b0 + 4);
    float4 a2 = *(const float4*)(p + b0 + 16);
    float4 a3 = *(const float4*)(p + b0 + 20);
    v16h f;
    f[0]=(_Float16)a0.x; f[1]=(_Float16)a0.y; f[2]=(_Float16)a0.z; f[3]=(_Float16)a0.w;
    f[4]=(_Float16)a1.x; f[5]=(_Float16)a1.y; f[6]=(_Float16)a1.z; f[7]=(_Float16)a1.w;
    f[8]=(_Float16)a2.x; f[9]=(_Float16)a2.y; f[10]=(_Float16)a2.z; f[11]=(_Float16)a2.w;
    f[12]=(_Float16)a3.x; f[13]=(_Float16)a3.y; f[14]=(_Float16)a3.z; f[15]=(_Float16)a3.w;
    return f;
}

// ---------------------------------------------------------------------------
// WMMA GEMM: C[M,N] = A[M,K] @ W[N,K]^T.  One wave computes a 16 x (16*NT)
// strip: the A fragment is loaded & converted once per k-step and reused for
// NT back-to-back wmma ops (fills the XDL pipe, 4x less A traffic for NT=4).
// EPI: 0 = plain store, 1 = +bias then ReLU, 2 = residual add (C += resid).
// ---------------------------------------------------------------------------
template <int EPI, int NT>
__global__ __launch_bounds__(32)
void wmma_gemm_kernel(const float* __restrict__ A, const float* __restrict__ W,
                      float* __restrict__ C,
                      const float* __restrict__ bias,
                      const float* __restrict__ resid,
                      int K, int N, int ldc)
{
    const int lane = threadIdx.x;       // 0..31
    const int half = lane >> 4;
    const int l16  = lane & 15;

    const long mrow  = (long)blockIdx.x * 16 + l16;
    const int  nbase = blockIdx.y * (16 * NT);

    const float* ap = A + mrow * (long)K;
    const float* wp[NT];
    #pragma unroll
    for (int nt = 0; nt < NT; ++nt) {
        int wr = nbase + nt * 16 + l16;
        wr = (wr < N) ? wr : (N - 1);   // branchless clamp for partial N tiles
        wp[nt] = W + (long)wr * (long)K;
    }

    v8f acc[NT] = {};
    for (int kk = 0; kk < K; kk += 32) {
        const int b0 = kk + half * 8;
        v16h af = load_frag_(ap, b0);
        #pragma unroll
        for (int nt = 0; nt < NT; ++nt) {
            v16h wf = load_frag_(wp[nt], b0);
            acc[nt] = __builtin_amdgcn_wmma_f32_16x16x32_f16(
                false, af, false, wf, (short)0, acc[nt], false, false);
        }
    }

    // D layout: VGPR r -> row (r + 8*half), col = l16
    const long cm0 = (long)blockIdx.x * 16 + half * 8;
    #pragma unroll
    for (int nt = 0; nt < NT; ++nt) {
        const int cn = nbase + nt * 16 + l16;
        if (cn < N) {
            #pragma unroll
            for (int r = 0; r < 8; ++r) {
                float v = acc[nt][r];
                long off = (cm0 + r) * (long)ldc + cn;
                if (EPI == 1) { v += bias[cn]; v = fmaxf(v, 0.0f); }
                if (EPI == 2) { v += resid[off]; }
                C[off] = v;
            }
        }
    }
}

// ---------------------------------------------------------------------------
// Branch input projection: act[b,t,d] = relu(sum_f x[b,t,fbase+f]*w[d,f]+b[d])
// ---------------------------------------------------------------------------
__global__ void branch_proj_kernel(const float* __restrict__ x,
                                   const float* __restrict__ w,
                                   const float* __restrict__ b,
                                   int fbase, int fcount,
                                   float* __restrict__ act)
{
    int i = blockIdx.x * blockDim.x + threadIdx.x;
    if (i >= BATCH * SEQ * DM) return;
    int d = i & (DM - 1);
    int row = i >> 6;
    const float* xp = x + (long)row * 10 + fbase;
    float s = b[d];
    for (int f = 0; f < fcount; ++f) s += xp[f] * w[d * fcount + f];
    act[i] = fmaxf(s, 0.0f);
}

// ---------------------------------------------------------------------------
// LayerNorm over last dim (64), one thread per row.
// ---------------------------------------------------------------------------
__global__ void layernorm_kernel(const float* __restrict__ act,
                                 const float* __restrict__ g,
                                 const float* __restrict__ bb,
                                 float* __restrict__ xn)
{
    int row = blockIdx.x * blockDim.x + threadIdx.x;
    if (row >= BATCH * SEQ) return;
    const float* p = act + (long)row * DM;
    float s = 0.0f, s2 = 0.0f;
    for (int d = 0; d < DM; ++d) { float v = p[d]; s += v; s2 += v * v; }
    float mu  = s * (1.0f / DM);
    float var = s2 * (1.0f / DM) - mu * mu;
    float inv = rsqrtf(var + 1e-5f);
    float* q = xn + (long)row * DM;
    for (int d = 0; d < DM; ++d) q[d] = (p[d] - mu) * inv * g[d] + bb[d];
}

// ---------------------------------------------------------------------------
// Depthwise causal conv (width 4) + bias + SiLU.  u lives in xz[..,0:128].
// ---------------------------------------------------------------------------
__global__ void conv_silu_kernel(const float* __restrict__ xz,
                                 const float* __restrict__ cw,
                                 const float* __restrict__ cb,
                                 float* __restrict__ uc)
{
    int i = blockIdx.x * blockDim.x + threadIdx.x;
    if (i >= BATCH * SEQ * DI) return;
    int d   = i & (DI - 1);
    int row = i >> 7;            // b*SEQ + t
    int t   = row & (SEQ - 1);
    int bL  = row - t;
    float s = cb[d];
    #pragma unroll
    for (int j = 0; j < 4; ++j) {
        int tt = t - 3 + j;
        if (tt >= 0) s += xz[((long)(bL + tt)) * 256 + d] * cw[d * 4 + j];
    }
    uc[i] = siluf_(s);
}

// ---------------------------------------------------------------------------
// Selective scan: one block = one batch element, one thread = one d_inner.
// Wave 0 issues ONE Tensor-Data-Mover load pulling the full (SEQ x 68) x_dbl
// tile (17.4 KB) into LDS, s_wait_tensorcnt, single barrier -- then the whole
// recurrence runs from LDS with no per-timestep barriers or global B/C loads.
// h[32] in registers; dt_proj + softplus fused in.
// ---------------------------------------------------------------------------
__global__ __launch_bounds__(128)
void scan_kernel(const float* __restrict__ xdbl, const float* __restrict__ uc,
                 const float* __restrict__ dtw, const float* __restrict__ dtb,
                 const float* __restrict__ Alog, const float* __restrict__ Dp,
                 float* __restrict__ y)
{
    __shared__ float sxd[SEQ * XDBL_W];     // 4352 floats = 17408 bytes
    const int d = threadIdx.x;
    const int b = blockIdx.x;

    if (threadIdx.x < 32) {                 // wave 0 issues the TDM copy
        unsigned long long gaddr =
            (unsigned long long)(const void*)(xdbl + (long)b * SEQ * XDBL_W);
        unsigned int laddr = (unsigned int)(unsigned long long)(void*)&sxd[0];
        const unsigned int elems = SEQ * XDBL_W;   // 4352 elements of 4 bytes

        v4u g0;
        g0[0] = 1u;                                        // count=1 (valid user D#)
        g0[1] = laddr;                                     // lds_addr
        g0[2] = (unsigned int)(gaddr & 0xffffffffu);       // global_addr[31:0]
        g0[3] = (unsigned int)((gaddr >> 32) & 0x01ffffffu)// global_addr[56:32]
              | (2u << 30);                                // type = 2 ("image")
        v8i g1;
        g1[0] = (int)(2u << 16);                 // workgroup_mask=0, data_size=4B
        g1[1] = (int)((elems & 0xffffu) << 16);  // tensor_dim0[15:0]
        g1[2] = (int)(((elems >> 16) & 0xffffu)) // tensor_dim0[31:16]
              | (1 << 16);                       // tensor_dim1 = 1
        g1[3] = (int)((elems & 0xffffu) << 16);  // tile_dim0 = elems
        g1[4] = 1;                               // tile_dim1 = 1, tile_dim2 = 0
        g1[5] = (int)elems;                      // tensor_dim0_stride[31:0]
        g1[6] = 0;
        g1[7] = 0;
        v4i g2 = {};                             // group2: unused (<=2D tensor)
        v4i g3 = {};                             // group3: unused
        v8i g4 = {};                             // extra operand (6-arg builtin)
        __builtin_amdgcn_tensor_load_to_lds(g0, g1, g2, g3, g4, 0);
        __builtin_amdgcn_s_wait_tensorcnt(0);
    }
    __syncthreads();

    float A[DS], h[DS];
    #pragma unroll
    for (int s = 0; s < DS; ++s) { A[s] = -__expf(Alog[d * DS + s]); h[s] = 0.0f; }
    const float w0 = dtw[d * 4 + 0], w1 = dtw[d * 4 + 1];
    const float w2 = dtw[d * 4 + 2], w3 = dtw[d * 4 + 3];
    const float db = dtb[d], Dd = Dp[d];

    for (int t = 0; t < SEQ; ++t) {
        const float* sx = &sxd[t * XDBL_W];
        const long row = (long)b * SEQ + t;
        float u  = uc[row * DI + d];
        float dt = softplusf_(sx[0]*w0 + sx[1]*w1 + sx[2]*w2 + sx[3]*w3 + db);
        float accv = 0.0f;
        #pragma unroll
        for (int s = 0; s < DS; ++s) {
            h[s] = h[s] * __expf(dt * A[s]) + dt * sx[4 + s] * u;
            accv += h[s] * sx[36 + s];
        }
        y[row * DI + d] = accv + u * Dd;
    }
}

// y *= silu(z);  z lives in xz[..,128:256]
__global__ void gate_silu_kernel(float* __restrict__ y, const float* __restrict__ xz)
{
    int i = blockIdx.x * blockDim.x + threadIdx.x;
    if (i >= BATCH * SEQ * DI) return;
    int d = i & (DI - 1);
    long row = i >> 7;
    float z = xz[row * 256 + DI + d];
    y[i] *= siluf_(z);
}

// mean over L into env[b, off + d]
__global__ void mean_kernel(const float* __restrict__ act, float* __restrict__ env, int off)
{
    int i = blockIdx.x * blockDim.x + threadIdx.x;
    if (i >= BATCH * DM) return;
    int d = i & (DM - 1);
    int b = i >> 6;
    float s = 0.0f;
    for (int t = 0; t < SEQ; ++t) s += act[((long)b * SEQ + t) * DM + d];
    env[b * 2 * DM + off + d] = s * (1.0f / SEQ);
}

// rain gate + fuse:  fused[b,j] = env[b,j] * (sigmoid(x_rain.lw + lb)*gw[j] + gb[j])
__global__ void fuse_kernel(const float* __restrict__ x, const float* __restrict__ env,
                            const float* __restrict__ lw, const float* __restrict__ lb,
                            const float* __restrict__ gw, const float* __restrict__ gb,
                            float* __restrict__ fused)
{
    int i = blockIdx.x * blockDim.x + threadIdx.x;
    if (i >= BATCH * 2 * DM) return;
    int j = i & (2 * DM - 1);
    int b = i >> 7;
    const float* xp = x + (long)b * SEQ * 10;   // t = 0, features 0..3
    float rp = sigmoidf_(xp[0]*lw[0] + xp[1]*lw[1] + xp[2]*lw[2] + xp[3]*lw[3] + lb[0]);
    fused[i] = env[i] * (rp * gw[j] + gb[j]);
}

// head2 dot + sigmoid + Tversky/BCE loss reduction (single block)
__global__ __launch_bounds__(256)
void head2_loss_kernel(const float* __restrict__ h1, const float* __restrict__ w2,
                       const float* __restrict__ b2, const int* __restrict__ target,
                       float* __restrict__ out)
{
    __shared__ float stp[256], sfn[256], sfp[256], sbce[256];
    int tid = threadIdx.x;
    float tp = 0, fn = 0, fp = 0, bce = 0;
    for (int b = tid; b < BATCH; b += 256) {
        const float* hp = h1 + (long)b * 64;
        float l = b2[0];
        for (int k = 0; k < 64; ++k) l += hp[k] * w2[k];
        float prob = sigmoidf_(l);
        out[b] = prob;
        float tf = (float)target[b];
        tp += tf * prob;
        fn += tf * (1.0f - prob);
        fp += (1.0f - tf) * prob;
        bce += 5.0f * tf * softplusf_(-l) + (1.0f - tf) * softplusf_(l);
    }
    stp[tid] = tp; sfn[tid] = fn; sfp[tid] = fp; sbce[tid] = bce;
    __syncthreads();
    for (int s = 128; s > 0; s >>= 1) {
        if (tid < s) {
            stp[tid] += stp[tid + s]; sfn[tid] += sfn[tid + s];
            sfp[tid] += sfp[tid + s]; sbce[tid] += sbce[tid + s];
        }
        __syncthreads();
    }
    if (tid == 0) {
        float l_ts  = -stp[0] / (stp[0] + sfn[0] + sfp[0] + 1e-6f);
        float l_bce = sbce[0] / (float)BATCH;
        out[BATCH] = 0.7f * l_ts + 0.3f * l_bce;
    }
}

// ---------------------------------------------------------------------------
extern "C" void kernel_launch(void* const* d_in, const int* in_sizes, int n_in,
                              void* d_out, int out_size, void* d_ws, size_t ws_size,
                              hipStream_t stream)
{
    (void)in_sizes; (void)n_in; (void)out_size; (void)ws_size;

    // Input layout: recursive dict-insertion-order flatten of setup_inputs().
    const float* x       = (const float*)d_in[0];
    const float* topo_w  = (const float*)d_in[1];
    const float* topo_b  = (const float*)d_in[2];
    const float* fuel_w  = (const float*)d_in[3];
    const float* fuel_b  = (const float*)d_in[4];
    const int TOPO_BASE = 5;    // 4 layers x 11 tensors
    const int FUEL_BASE = 49;
    const float* logit_w = (const float*)d_in[93];
    const float* logit_b = (const float*)d_in[94];
    const float* gate_w  = (const float*)d_in[95];
    const float* gate_b  = (const float*)d_in[96];
    const float* head1_w = (const float*)d_in[97];
    const float* head1_b = (const float*)d_in[98];
    const float* head2_w = (const float*)d_in[99];
    const float* head2_b = (const float*)d_in[100];
    const int*   target  = (const int*)d_in[101];
    float* out = (float*)d_out;

    // Workspace carve-up (floats). Total ~23.4M floats (~94 MB) — fits in L2.
    float* ws = (float*)d_ws;
    size_t o = 0;
    float* act  = ws + o; o += (size_t)BATCH * SEQ * DM;
    float* xn   = ws + o; o += (size_t)BATCH * SEQ * DM;
    float* xz   = ws + o; o += (size_t)BATCH * SEQ * 2 * DI;
    float* uc   = ws + o; o += (size_t)BATCH * SEQ * DI;
    float* xdbl = ws + o; o += (size_t)BATCH * SEQ * XDBL_W;
    float* ybuf = ws + o; o += (size_t)BATCH * SEQ * DI;
    float* env  = ws + o; o += (size_t)BATCH * 2 * DM;
    float* fusd = ws + o; o += (size_t)BATCH * 2 * DM;
    float* h1   = ws + o; o += (size_t)BATCH * 64;

    const int ROWS = BATCH * SEQ;   // 32768 (multiple of 16)

    for (int br = 0; br < 2; ++br) {
        const float* pw = (br == 0) ? topo_w : fuel_w;
        const float* pb = (br == 0) ? topo_b : fuel_b;
        const int fbase = (br == 0) ? 4 : 5;
        const int fcnt  = (br == 0) ? 1 : 5;
        const int base  = (br == 0) ? TOPO_BASE : FUEL_BASE;

        branch_proj_kernel<<<(ROWS * DM + 255) / 256, 256, 0, stream>>>(
            x, pw, pb, fbase, fcnt, act);

        for (int li = 0; li < 4; ++li) {
            const float* in_proj_w  = (const float*)d_in[base + li * 11 + 0];
            const float* conv_w     = (const float*)d_in[base + li * 11 + 1];
            const float* conv_b     = (const float*)d_in[base + li * 11 + 2];
            const float* x_proj_w   = (const float*)d_in[base + li * 11 + 3];
            const float* dt_proj_w  = (const float*)d_in[base + li * 11 + 4];
            const float* dt_proj_b  = (const float*)d_in[base + li * 11 + 5];
            const float* A_log      = (const float*)d_in[base + li * 11 + 6];
            const float* Dp         = (const float*)d_in[base + li * 11 + 7];
            const float* out_proj_w = (const float*)d_in[base + li * 11 + 8];
            const float* ln_g       = (const float*)d_in[base + li * 11 + 9];
            const float* ln_b       = (const float*)d_in[base + li * 11 + 10];

            layernorm_kernel<<<(ROWS + 255) / 256, 256, 0, stream>>>(act, ln_g, ln_b, xn);

            // xz = xn @ in_proj_w^T   (32768x64 @ 64x256), 16x64 strip per wave
            wmma_gemm_kernel<0, 4><<<dim3(ROWS / 16, 4), 32, 0, stream>>>(
                xn, in_proj_w, xz, nullptr, nullptr, DM, 2 * DI, 2 * DI);

            conv_silu_kernel<<<(ROWS * DI + 255) / 256, 256, 0, stream>>>(
                xz, conv_w, conv_b, uc);

            // xdbl = uc @ x_proj_w^T  (32768x128 @ 128x68), guarded N tiles
            wmma_gemm_kernel<0, 1><<<dim3(ROWS / 16, (XDBL_W + 15) / 16), 32, 0, stream>>>(
                uc, x_proj_w, xdbl, nullptr, nullptr, DI, XDBL_W, XDBL_W);

            scan_kernel<<<BATCH, DI, 0, stream>>>(
                xdbl, uc, dt_proj_w, dt_proj_b, A_log, Dp, ybuf);

            gate_silu_kernel<<<(ROWS * DI + 255) / 256, 256, 0, stream>>>(ybuf, xz);

            // act += ybuf @ out_proj_w^T   (residual), 16x64 strip per wave
            wmma_gemm_kernel<2, 4><<<dim3(ROWS / 16, 1), 32, 0, stream>>>(
                ybuf, out_proj_w, act, nullptr, act, DI, DM, DM);
        }
        mean_kernel<<<(BATCH * DM + 255) / 256, 256, 0, stream>>>(act, env, br * DM);
    }

    fuse_kernel<<<(BATCH * 2 * DM + 255) / 256, 256, 0, stream>>>(
        x, env, logit_w, logit_b, gate_w, gate_b, fusd);

    // h1 = relu(fused @ head1_w^T + head1_b)   (512x128 @ 128x64)
    wmma_gemm_kernel<1, 4><<<dim3(BATCH / 16, 1), 32, 0, stream>>>(
        fusd, head1_w, h1, head1_b, nullptr, 2 * DM, DM, DM);

    head2_loss_kernel<<<1, 256, 0, stream>>>(h1, head2_w, head2_b, target, out);
}
